// EnhancedGAT_2637109920383
// MI455X (gfx1250) — matrix operand order repaired and checked by reading, not compiled, not run on hardware
//
#include <hip/hip_runtime.h>
#include <hip/hip_bf16.h>

// ---------------------------------------------------------------------------
// EnhancedGAT on gfx1250 (MI455X, wave32).
//   N=50000 nodes, E=800000 edges (+N self loops = 850000), IN=128,
//   layer1: H=4,C=64 (concat->256), layer2: H=1,C=64, proj->128.
// GEMMs: v_wmma_f32_16x16x32_f16, B panel staged to LDS by the Tensor Data
// Mover (tensor_load_to_lds + s_wait_tensorcnt). Edge softmax/scatter in fp32
// with global atomics (h1 51.2MB + accum 51.2MB both L2-resident, 192MB L2).
// Biases cancel through the following BatchNorm mean-subtraction -> skipped.
// ---------------------------------------------------------------------------

typedef __attribute__((ext_vector_type(16))) _Float16     v16h;
typedef __attribute__((ext_vector_type(8)))  float        v8f;
typedef __attribute__((ext_vector_type(4)))  unsigned int u32x4;
typedef __attribute__((ext_vector_type(8)))  int          i32x8;
typedef __attribute__((ext_vector_type(4)))  int          i32x4;

#define GAT_N   50000
#define GAT_E   800000
#define GAT_EP  850000      /* E + N self loops */
#define GAT_IN  128
#define GAT_F1  256         /* H1*C1 */
#define GAT_H1  4
#define GAT_F2  64          /* H2*C2 */
#define GAT_H2  1

// ---------------------------------------------------------------------------
// WMMA GEMM: C[M,Nd] = A[M,K] * B[K,Nd], fp32 in/out, f16 multiply, f32 acc.
// Block = 256 thr (8 waves) = one 16-col N tile x eight 16-row M tiles.
// B panel (K x 16 fp32) is DMA'd global->LDS once per block by the TDM with
// 1-DWORD padding every 16 DWORDs (row stride 17 words -> bank-window split
// between the two 16-lane halves). Waves read B frags from LDS, A frags as
// contiguous global_load_b128. Requires M%16==0, K%32==0, Nd%16==0, K<=256.
// Fragment layouts per CDNA5 ISA 7.12.2.
// ---------------------------------------------------------------------------
__global__ void gat_gemm_wmma_f16(const float* __restrict__ A,
                                  const float* __restrict__ B,
                                  float* __restrict__ C,
                                  int M, int K, int Nd) {
  __shared__ float bpanel[256 * 17];          // K<=256 rows, 17-word stride

  const int ntiles = Nd >> 4;
  const int mtiles = M >> 4;
  const int nt   = blockIdx.x % ntiles;
  const int mblk = blockIdx.x / ntiles;

  // ---- stage B[0..K-1, nt*16 .. nt*16+15] into LDS via Tensor Data Mover ----
  if (threadIdx.x < 32) {
    unsigned long long gaddr =
        (unsigned long long)(const char*)(B + (size_t)nt * 16);
    unsigned int ldsaddr = (unsigned int)(size_t)(&bpanel[0]);

    u32x4 g0;
    g0[0] = 1u;                                   // count=1, user mode
    g0[1] = ldsaddr;                              // lds_addr
    g0[2] = (unsigned int)(gaddr & 0xFFFFFFFFu);  // global_addr[31:0]
    g0[3] = (unsigned int)((gaddr >> 32) & 0x1FFFFFFu) | 0x80000000u; // +type=2

    i32x8 g1;
    g1[0] = (2 << 16) | (1 << 20) | (3 << 22);    // 4B elems, pad_en, every 16 DW, +1 DW
    g1[1] = (int)((Nd & 0xFFFF) << 16);           // tensor_dim0 lo16
    g1[2] = (int)(((unsigned)Nd >> 16) | ((K & 0xFFFF) << 16)); // dim0 hi | dim1 lo
    g1[3] = (int)(((unsigned)K >> 16) | (16 << 16));            // dim1 hi | tile_dim0=16
    g1[4] = (int)(K & 0xFFFF);                    // tile_dim1=K, tile_dim2=0
    g1[5] = (int)Nd;                              // tensor_dim0_stride lo32
    g1[6] = 0;
    g1[7] = 0;

    i32x4 z4 = {0, 0, 0, 0};
#if defined(__clang_major__) && (__clang_major__ >= 23)
    i32x8 z8 = {0, 0, 0, 0, 0, 0, 0, 0};
    __builtin_amdgcn_tensor_load_to_lds(g0, g1, z4, z4, z8, 0);
#else
    __builtin_amdgcn_tensor_load_to_lds(g0, g1, z4, z4, 0);
#endif
    __builtin_amdgcn_s_wait_tensorcnt(0);
  }
  __syncthreads();

  const int mt = mblk * 8 + (threadIdx.x >> 5);
  if (mt >= mtiles) return;

  const int lane = threadIdx.x & 31;
  const int hs   = lane >> 4;          // half select
  const int l    = lane & 15;

  const float* __restrict__ arow = A + (size_t)(mt * 16 + l) * K;

  v8f acc = {};
  for (int k0 = 0; k0 < K; k0 += 32) {
    v16h a, b;
    const float* ap0 = arow + k0 + hs * 8;        // K = k0 + hs*8 + j
    const float* ap1 = arow + k0 + 16 + hs * 8;   // K = k0+16 + hs*8 + j
#pragma unroll
    for (int j = 0; j < 8; ++j) {
      a[j]     = (_Float16)ap0[j];
      a[8 + j] = (_Float16)ap1[j];
    }
    const float* bp = &bpanel[(k0 + hs * 8) * 17 + l];
#pragma unroll
    for (int j = 0; j < 8; ++j) { b[j] = (_Float16)bp[j * 17]; b[8 + j] = (_Float16)bp[(16 + j) * 17]; }

    if (k0 + 32 < K)   // warm caches for next A k-block -> global_prefetch_b8
      __builtin_prefetch(arow + k0 + 32, 0, 1);

    acc = __builtin_amdgcn_wmma_f32_16x16x32_f16(
        /*neg_a=*/false, a, /*neg_b=*/false, b,
        /*c_mod=*/(short)0, acc, /*reuse_a=*/false, /*reuse_b=*/false);
  }

  float* crow = C + (size_t)(mt * 16 + hs * 8) * Nd + nt * 16 + l;
#pragma unroll
  for (int i = 0; i < 8; ++i) crow[(size_t)i * Nd] = acc[i];
}

// ---------------------------------------------------------------------------
// alpha_{s,d}[n,h] = sum_c h[n,h,c] * a_{s,d}[h,c]   (C fixed = 64)
// one wave per (node, head), shuffle tree reduce.
// ---------------------------------------------------------------------------
__global__ void gat_alpha_kernel(const float* __restrict__ h,
                                 const float* __restrict__ a_src,
                                 const float* __restrict__ a_dst,
                                 float* __restrict__ as, float* __restrict__ ad,
                                 int Nn, int Hh) {
  const int pair = blockIdx.x * (blockDim.x >> 5) + (threadIdx.x >> 5);
  if (pair >= Nn * Hh) return;
  const int n  = pair / Hh;
  const int hd = pair - n * Hh;
  const int lane = threadIdx.x & 31;

  const float* hp  = h + ((size_t)n * Hh + hd) * 64;
  const float* asp = a_src + hd * 64;
  const float* adp = a_dst + hd * 64;
  float v0 = hp[lane], v1 = hp[lane + 32];
  float s0 = v0 * asp[lane] + v1 * asp[lane + 32];
  float s1 = v0 * adp[lane] + v1 * adp[lane + 32];
#pragma unroll
  for (int off = 16; off > 0; off >>= 1) {
    s0 += __shfl_xor(s0, off, 32);
    s1 += __shfl_xor(s1, off, 32);
  }
  if (lane == 0) { as[pair] = s0; ad[pair] = s1; }
}

// monotonic float<->uint key for atomicMax-based segment max
__device__ __forceinline__ unsigned fkey(float f) {
  unsigned u = __float_as_uint(f);
  return (u & 0x80000000u) ? ~u : (u | 0x80000000u);
}
__device__ __forceinline__ float kval(unsigned k) {
  unsigned u = (k & 0x80000000u) ? (k & 0x7FFFFFFFu) : ~k;
  return __uint_as_float(u);
}
__device__ __forceinline__ float lrelu(float v) { return v > 0.f ? v : 0.2f * v; }

// segment max over destinations (thread per edge; self loops synthesized)
__global__ void gat_edge_max(const int* __restrict__ ei,
                             const float* __restrict__ as,
                             const float* __restrict__ ad,
                             unsigned* __restrict__ mk, int Hh) {
  const int e = blockIdx.x * blockDim.x + threadIdx.x;
  if (e >= GAT_EP) return;
  const int s = (e < GAT_E) ? ei[e]         : (e - GAT_E);
  const int d = (e < GAT_E) ? ei[GAT_E + e] : (e - GAT_E);
  for (int hh = 0; hh < Hh; ++hh) {
    float v = lrelu(as[s * Hh + hh] + ad[d * Hh + hh]);
    atomicMax(&mk[d * Hh + hh], fkey(v));
  }
}

// ex = exp(e - m[dst]); den[dst] += ex
__global__ void gat_edge_expsum(const int* __restrict__ ei,
                                const float* __restrict__ as,
                                const float* __restrict__ ad,
                                const unsigned* __restrict__ mk,
                                float* __restrict__ ex, float* __restrict__ den,
                                int Hh) {
  const int e = blockIdx.x * blockDim.x + threadIdx.x;
  if (e >= GAT_EP) return;
  const int s = (e < GAT_E) ? ei[e]         : (e - GAT_E);
  const int d = (e < GAT_E) ? ei[GAT_E + e] : (e - GAT_E);
  for (int hh = 0; hh < Hh; ++hh) {
    float v = lrelu(as[s * Hh + hh] + ad[d * Hh + hh]);
    float m = kval(mk[d * Hh + hh]);   // every node has a self loop -> valid
    float x = __expf(v - m);
    ex[(size_t)e * Hh + hh] = x;
    atomicAdd(&den[d * Hh + hh], x);
  }
}

// out[dst,h,:] += (ex/den) * h[src,h,:]   (wave per edge, lanes over 64 ch)
__global__ void gat_edge_scatter(const int* __restrict__ ei,
                                 const float* __restrict__ h,
                                 const float* __restrict__ ex,
                                 const float* __restrict__ den,
                                 float* __restrict__ out, int Hh) {
  const int e = blockIdx.x * (blockDim.x >> 5) + (threadIdx.x >> 5);
  if (e >= GAT_EP) return;
  const int lane = threadIdx.x & 31;
  const int s = (e < GAT_E) ? ei[e]         : (e - GAT_E);
  const int d = (e < GAT_E) ? ei[GAT_E + e] : (e - GAT_E);
  const int F = Hh * 64;
  for (int hh = 0; hh < Hh; ++hh) {
    float w = ex[(size_t)e * Hh + hh] / (den[d * Hh + hh] + 1e-16f);
    const float* hp = h + (size_t)s * F + hh * 64;
    float* op       = out + (size_t)d * F + hh * 64;
    atomicAdd(&op[lane],      w * hp[lane]);
    atomicAdd(&op[lane + 32], w * hp[lane + 32]);
  }
}

// column partial sums (coalesced: thread -> column = tid % F)
__global__ void gat_colsum(const float* __restrict__ X, float* __restrict__ gsum,
                           float* __restrict__ gsq, int Nn, int F, int rowsPerBlock) {
  const int f     = threadIdx.x % F;
  const int rstep = blockDim.x / F;
  int r    = blockIdx.x * rowsPerBlock + threadIdx.x / F;
  int rend = min(Nn, (blockIdx.x + 1) * rowsPerBlock);
  float s = 0.f, q = 0.f;
  for (; r < rend; r += rstep) {
    float v = X[(size_t)r * F + f];
    s += v; q += v * v;
  }
  atomicAdd(&gsum[f], s);
  atomicAdd(&gsq[f], q);
}

__global__ void gat_stats_finalize(const float* __restrict__ gsum,
                                   const float* __restrict__ gsq,
                                   float* __restrict__ mean, float* __restrict__ rstd,
                                   int Nn, int F) {
  const int f = blockIdx.x * blockDim.x + threadIdx.x;
  if (f >= F) return;
  float mu  = gsum[f] / (float)Nn;
  float var = gsq[f] / (float)Nn - mu * mu;
  mean[f] = mu;
  rstd[f] = rsqrtf(var + 1e-5f);
}

// y = [elu]((x - mu) * rstd * gamma + beta); in-place safe
__global__ void gat_bn_apply(const float* __restrict__ X, float* __restrict__ Y,
                             const float* __restrict__ mean, const float* __restrict__ rstd,
                             const float* __restrict__ g, const float* __restrict__ be,
                             int Nn, int F, int doElu) {
  const size_t i = (size_t)blockIdx.x * blockDim.x + threadIdx.x;
  if (i >= (size_t)Nn * F) return;
  const int f = (int)(i % F);
  float v = (X[i] - mean[f]) * rstd[f] * g[f] + be[f];
  if (doElu) v = v > 0.f ? v : (__expf(v) - 1.f);
  Y[i] = v;
}

static inline int ceil_div_i(long long a, long long b) { return (int)((a + b - 1) / b); }

extern "C" void kernel_launch(void* const* d_in, const int* in_sizes, int n_in,
                              void* d_out, int out_size, void* d_ws, size_t ws_size,
                              hipStream_t stream) {
  const float* x      = (const float*)d_in[0];
  const int*   ei     = (const int*)d_in[1];
  const float* W1     = (const float*)d_in[2];
  const float* a1_src = (const float*)d_in[3];
  const float* a1_dst = (const float*)d_in[4];
  const float* W2     = (const float*)d_in[6];
  const float* a2_src = (const float*)d_in[7];
  const float* a2_dst = (const float*)d_in[8];
  const float* Wp     = (const float*)d_in[10];
  const float* g1  = (const float*)d_in[12];
  const float* be1 = (const float*)d_in[13];
  const float* g2  = (const float*)d_in[14];
  const float* be2 = (const float*)d_in[15];
  const float* g3  = (const float*)d_in[16];
  const float* be3 = (const float*)d_in[17];
  float* out = (float*)d_out;

  // ---- workspace layout (bump, with reuse); ~119 MB total ----
  char* w = (char*)d_ws;
  const size_t szA = (size_t)GAT_N * GAT_F1 * 4;   // 51.2 MB
  float* h1    = (float*)w;                         // layer1 features
  // layer2/3 sub-allocations reuse h1's span after it dies (post scatter1):
  float* h2    = (float*)w;                         // N*64
  float* g2out = (float*)(w + (size_t)GAT_N * GAT_F2 * 4);        // N*64
  float* proj  = (float*)(w + (size_t)2 * GAT_N * GAT_F2 * 4);    // N*128
  w += szA;
  float* g1out = (float*)w; w += szA;               // layer1 accum -> h1act (in-place BN)
  float* exbuf = (float*)w; w += (size_t)GAT_EP * GAT_H1 * 4;     // 13.6 MB, reused layer2
  float* as1   = (float*)w; w += (size_t)GAT_N * GAT_H1 * 4;
  float* ad1   = (float*)w; w += (size_t)GAT_N * GAT_H1 * 4;
  unsigned* mk = (unsigned*)w; w += (size_t)GAT_N * GAT_H1 * 4;   // reused layer2
  float* den   = (float*)w; w += (size_t)GAT_N * GAT_H1 * 4;      // reused layer2
  float* gsum  = (float*)w; w += 256 * 4;
  float* gsq   = (float*)w; w += 256 * 4;
  float* meanb = (float*)w; w += 256 * 4;
  float* rstdb = (float*)w; w += 256 * 4;

  const dim3 B256(256);
  const int mblks = ceil_div_i(GAT_N / 16, 8);     // 391 M-blocks of 8 tiles

  // ================= layer 1 =================
  hipMemsetAsync(mk,    0, (size_t)GAT_N * GAT_H1 * 4, stream);
  hipMemsetAsync(den,   0, (size_t)GAT_N * GAT_H1 * 4, stream);
  hipMemsetAsync(g1out, 0, szA, stream);

  // h1 = x @ W1   [50000,128]x[128,256]
  gat_gemm_wmma_f16<<<(GAT_F1 / 16) * mblks, B256, 0, stream>>>(
      x, W1, h1, GAT_N, GAT_IN, GAT_F1);
  gat_alpha_kernel<<<ceil_div_i((long long)GAT_N * GAT_H1, 8), B256, 0, stream>>>(
      h1, a1_src, a1_dst, as1, ad1, GAT_N, GAT_H1);
  gat_edge_max<<<ceil_div_i(GAT_EP, 256), B256, 0, stream>>>(ei, as1, ad1, mk, GAT_H1);
  gat_edge_expsum<<<ceil_div_i(GAT_EP, 256), B256, 0, stream>>>(ei, as1, ad1, mk, exbuf, den, GAT_H1);
  gat_edge_scatter<<<ceil_div_i(GAT_EP, 8), B256, 0, stream>>>(ei, h1, exbuf, den, g1out, GAT_H1);

  hipMemsetAsync(gsum, 0, GAT_F1 * 4, stream);
  hipMemsetAsync(gsq,  0, GAT_F1 * 4, stream);
  gat_colsum<<<ceil_div_i(GAT_N, 256), B256, 0, stream>>>(g1out, gsum, gsq, GAT_N, GAT_F1, 256);
  gat_stats_finalize<<<1, B256, 0, stream>>>(gsum, gsq, meanb, rstdb, GAT_N, GAT_F1);
  gat_bn_apply<<<ceil_div_i((long long)GAT_N * GAT_F1, 256), B256, 0, stream>>>(
      g1out, g1out, meanb, rstdb, g1, be1, GAT_N, GAT_F1, 1);   // in-place -> h1act

  // ================= layer 2 =================
  hipMemsetAsync(mk,  0, (size_t)GAT_N * GAT_H2 * 4, stream);
  hipMemsetAsync(den, 0, (size_t)GAT_N * GAT_H2 * 4, stream);
  hipMemsetAsync(g2out, 0, (size_t)GAT_N * GAT_F2 * 4, stream);

  // h2 = h1act @ W2   [50000,256]x[256,64]  (h2 lives where h1 was; h1 is dead)
  gat_gemm_wmma_f16<<<(GAT_F2 / 16) * mblks, B256, 0, stream>>>(
      g1out, W2, h2, GAT_N, GAT_F1, GAT_F2);
  gat_alpha_kernel<<<ceil_div_i((long long)GAT_N * GAT_H2, 8), B256, 0, stream>>>(
      h2, a2_src, a2_dst, as1, ad1, GAT_N, GAT_H2);
  gat_edge_max<<<ceil_div_i(GAT_EP, 256), B256, 0, stream>>>(ei, as1, ad1, mk, GAT_H2);
  gat_edge_expsum<<<ceil_div_i(GAT_EP, 256), B256, 0, stream>>>(ei, as1, ad1, mk, exbuf, den, GAT_H2);
  gat_edge_scatter<<<ceil_div_i(GAT_EP, 8), B256, 0, stream>>>(ei, h2, exbuf, den, g2out, GAT_H2);

  hipMemsetAsync(gsum, 0, GAT_F2 * 4, stream);
  hipMemsetAsync(gsq,  0, GAT_F2 * 4, stream);
  gat_colsum<<<ceil_div_i(GAT_N, 256), B256, 0, stream>>>(g2out, gsum, gsq, GAT_N, GAT_F2, 256);
  gat_stats_finalize<<<1, B256, 0, stream>>>(gsum, gsq, meanb, rstdb, GAT_N, GAT_F2);
  gat_bn_apply<<<ceil_div_i((long long)GAT_N * GAT_F2, 256), B256, 0, stream>>>(
      g2out, g2out, meanb, rstdb, g2, be2, GAT_N, GAT_F2, 1);

  // ================= projection + final BN =================
  // proj = h2act @ Wp   [50000,64]x[64,128]
  gat_gemm_wmma_f16<<<(GAT_IN / 16) * mblks, B256, 0, stream>>>(
      g2out, Wp, proj, GAT_N, GAT_F2, GAT_IN);
  hipMemsetAsync(gsum, 0, GAT_IN * 4, stream);
  hipMemsetAsync(gsq,  0, GAT_IN * 4, stream);
  gat_colsum<<<ceil_div_i(GAT_N, 256), B256, 0, stream>>>(proj, gsum, gsq, GAT_N, GAT_IN, 256);
  gat_stats_finalize<<<1, B256, 0, stream>>>(gsum, gsq, meanb, rstdb, GAT_N, GAT_IN);
  gat_bn_apply<<<ceil_div_i((long long)GAT_N * GAT_IN, 256), B256, 0, stream>>>(
      proj, out, meanb, rstdb, g3, be3, GAT_N, GAT_IN, 0);   // no ELU on final BN

  (void)in_sizes; (void)n_in; (void)out_size; (void)ws_size;
}
